// Encoder_2310692405382
// MI455X (gfx1250) — matrix operand order repaired and checked by reading, not compiled, not run on hardware
//
#include <hip/hip_runtime.h>

#define NNODES 50000
#define HID 256

typedef __attribute__((ext_vector_type(16))) __bf16 v16bf;
typedef __attribute__((ext_vector_type(2)))  __bf16 v2bf;
typedef __attribute__((ext_vector_type(8)))  float  v8f;

union Frag16 { v16bf v; uint4 q[2]; };

// native f32->bf16 (RNE) convert + pack; lowers to hardware cvt on gfx1250
__device__ __forceinline__ unsigned int pk2bf(float a, float b) {
  v2bf p;
  p.x = (__bf16)a;
  p.y = (__bf16)b;
  return __builtin_bit_cast(unsigned int, p);
}

// C[M x 256] = A[M x K](f32) * W[K x 256](f32); bf16 WMMA, f32 accumulate.
// Block: 256 thr (8 waves), tile 64(M) x 256(N). Wave: 2 M-tiles x 4 N-tiles.
// Double-buffered LDS staging with on-the-fly f32->bf16 conversion.
__global__ void __launch_bounds__(256) gemm_bf16wmma(
    const float* __restrict__ A, const float* __restrict__ W,
    float* __restrict__ C, int M, int K) {
  // row strides are 16B multiples (b128-friendly) and spread banks per half-wave
  __shared__ unsigned short As[2][64][40];    // 64x32 bf16 tile, 20 dwords/row
  __shared__ unsigned short Bs[2][32][264];   // 32x256 bf16 tile, 132 dwords/row

  const int tid  = threadIdx.x;
  const int m0   = blockIdx.x * 64;
  const int w    = tid >> 5;
  const int lane = tid & 31;
  const int half = lane >> 4;
  const int l16  = lane & 15;
  const int mg   = w & 1;        // 2 row-groups of 32
  const int ng   = w >> 1;       // 4 col-groups of 64

  v8f zero = {0.f,0.f,0.f,0.f,0.f,0.f,0.f,0.f};
  v8f acc[2][4];
#pragma unroll
  for (int mi = 0; mi < 2; ++mi)
#pragma unroll
    for (int t = 0; t < 4; ++t) acc[mi][t] = zero;

  // staging assignments
  const int ar = tid >> 2, acg = tid & 3;   // A: 64 rows x (4 grp * 8 floats)
  const int br = tid >> 3, bcg = tid & 7;   // B: 32 rows x (8 grp * 32 floats)
  const int agrow = m0 + ar;

  float4 a0f, a1f, bf[8];
  const int nsteps = K >> 5;

  // ---- load tile 0 ----
  {
    a0f = {}; a1f = {};
    if (agrow < M) {
      const float4* p = (const float4*)(A + (size_t)agrow * K + 8 * acg);
      a0f = p[0]; a1f = p[1];
    }
    const float4* q = (const float4*)(W + (size_t)br * 256 + 32 * bcg);
#pragma unroll
    for (int j = 0; j < 8; ++j) bf[j] = q[j];
  }

  for (int i = 0; i < nsteps; ++i) {
    const int buf = i & 1;
    // ---- convert + store staged tile into LDS[buf] ----
    {
      uint4 av;
      av.x = pk2bf(a0f.x, a0f.y); av.y = pk2bf(a0f.z, a0f.w);
      av.z = pk2bf(a1f.x, a1f.y); av.w = pk2bf(a1f.z, a1f.w);
      *(uint4*)&As[buf][ar][8 * acg] = av;
#pragma unroll
      for (int j = 0; j < 4; ++j) {
        uint4 bv;
        bv.x = pk2bf(bf[2*j].x, bf[2*j].y);     bv.y = pk2bf(bf[2*j].z, bf[2*j].w);
        bv.z = pk2bf(bf[2*j+1].x, bf[2*j+1].y); bv.w = pk2bf(bf[2*j+1].z, bf[2*j+1].w);
        *(uint4*)&Bs[buf][br][32 * bcg + 8 * j] = bv;
      }
    }
    __syncthreads();

    // ---- issue global loads for tile i+1 (overlap with compute) ----
    if (i + 1 < nsteps) {
      const int k0 = 32 * (i + 1);
      a0f = {}; a1f = {};
      if (agrow < M) {
        const float4* p = (const float4*)(A + (size_t)agrow * K + k0 + 8 * acg);
        a0f = p[0]; a1f = p[1];
      }
      const float4* q = (const float4*)(W + (size_t)(k0 + br) * 256 + 32 * bcg);
#pragma unroll
      for (int j = 0; j < 8; ++j) bf[j] = q[j];
    }

    // ---- compute from LDS[buf] ----
    // A frag (16-bit 16x32 ISA layout): lane<16 K{0..7,16..23}, lane>=16 K{8..15,24..31}
    Frag16 a0, a1;
    const int mr0 = mg * 32 + l16;
    a0.q[0] = *(const uint4*)&As[buf][mr0][8 * half];
    a0.q[1] = *(const uint4*)&As[buf][mr0][16 + 8 * half];
    a1.q[0] = *(const uint4*)&As[buf][mr0 + 16][8 * half];
    a1.q[1] = *(const uint4*)&As[buf][mr0 + 16][16 + 8 * half];
#pragma unroll
    for (int t = 0; t < 4; ++t) {
      // B frag: lane = N column (l16), K{0..7,16..23} for lane<16, K{8..15,24..31} for lane>=16
      Frag16 b;
      const int n0 = ng * 64 + 16 * t;
#pragma unroll
      for (int s = 0; s < 16; ++s)
        b.v[s] = __builtin_bit_cast(__bf16, Bs[buf][(s < 8 ? 8 * half + s : 8 + 8 * half + s)][n0 + l16]);
      acc[0][t] = __builtin_amdgcn_wmma_f32_16x16x32_bf16(
          false, a0.v, false, b.v, (short)0, acc[0][t], false, false);
      acc[1][t] = __builtin_amdgcn_wmma_f32_16x16x32_bf16(
          false, a1.v, false, b.v, (short)0, acc[1][t], false, false);
    }
    __syncthreads();
  }

  // D layout: elem r -> M = r + 8*half, N = lane&15
#pragma unroll
  for (int mi = 0; mi < 2; ++mi) {
    const int gm = m0 + mg * 32 + 16 * mi + 8 * half;
#pragma unroll
    for (int t = 0; t < 4; ++t) {
      const int gn = ng * 64 + 16 * t + l16;
#pragma unroll
      for (int r = 0; r < 8; ++r) {
        const int row = gm + r;
        if (row < M) C[(size_t)row * 256 + gn] = acc[mi][t][r];
      }
    }
  }
}

__global__ void deg_init(float* deg, int n) {
  int i = blockIdx.x * blockDim.x + threadIdx.x;
  if (i < n) deg[i] = 1.0f;                     // self-loop weight
}
__global__ void deg_acc(float* deg, const int* __restrict__ dst,
                        const float* __restrict__ ew, int nE) {
  int e = blockIdx.x * blockDim.x + threadIdx.x;
  if (e < nE) atomicAdd(deg + dst[e], ew[e]);
}
__global__ void deg_rsqrt(float* deg, int n) {
  int i = blockIdx.x * blockDim.x + threadIdx.x;
  if (i < n) { float d = deg[i]; deg[i] = d > 0.f ? rsqrtf(d) : 0.f; }
}
__global__ void norm_kernel(const float* __restrict__ dis,
                            const int* __restrict__ src, const int* __restrict__ dst,
                            const float* __restrict__ ew, float* __restrict__ norm,
                            int nE, int total) {
  int e = blockIdx.x * blockDim.x + threadIdx.x;
  if (e >= total) return;
  int s, d; float w;
  if (e < nE) { s = src[e]; d = dst[e]; w = ew[e]; }
  else        { s = d = e - nE; w = 1.f; }
  norm[e] = dis[s] * w * dis[d];
}
__global__ void bias_init(float* __restrict__ buf, const float* __restrict__ b, int n) {
  int i = blockIdx.x * blockDim.x + threadIdx.x;
  if (i < n) buf[i] = b[i & (HID - 1)];
}
// one wave per edge; lane handles channels lane + 32*j (coalesced, L2-resident)
__global__ void __launch_bounds__(256) prop_kernel(
    const float* __restrict__ h, const int* __restrict__ src,
    const int* __restrict__ dst, const float* __restrict__ norm,
    float* __restrict__ agg, int nE, int total) {
  int e = blockIdx.x * 8 + (threadIdx.x >> 5);
  if (e >= total) return;
  int lane = threadIdx.x & 31;
  int s, d;
  if (e < nE) { s = src[e]; d = dst[e]; }
  else        { s = d = e - nE; }
  float w = norm[e];
  const float* hp = h + (size_t)s * HID;
  float* ap = agg + (size_t)d * HID;
#pragma unroll
  for (int j = 0; j < 8; ++j) {
    int c = lane + 32 * j;
    atomicAdd(ap + c, hp[c] * w);
  }
}
__global__ void prelu_kernel(float* __restrict__ out, const float* __restrict__ a, int n) {
  int i = blockIdx.x * blockDim.x + threadIdx.x;
  if (i < n) {
    float h = out[i];
    out[i] = fmaxf(h, 0.f) + a[i & (HID - 1)] * fminf(h, 0.f);
  }
}

extern "C" void kernel_launch(void* const* d_in, const int* in_sizes, int n_in,
                              void* d_out, int out_size, void* d_ws, size_t ws_size,
                              hipStream_t stream) {
  const float* x  = (const float*)d_in[0];
  const int*   ei = (const int*)d_in[1];
  const float* ew = (const float*)d_in[2];
  const float* W1 = (const float*)d_in[3];  const float* b1 = (const float*)d_in[4];
  const float* W2 = (const float*)d_in[5];  const float* b2 = (const float*)d_in[6];
  const float* W3 = (const float*)d_in[7];  const float* b3 = (const float*)d_in[8];
  const float* W4 = (const float*)d_in[9];  const float* b4 = (const float*)d_in[10];
  const float* pa = (const float*)d_in[11];
  float* out = (float*)d_out;

  const int nE    = in_sizes[2];            // 800000
  const int N     = NNODES;
  const int K1    = in_sizes[0] / N;        // 512
  const int total = nE + N;                 // edges + self-loops
  const int* srcp = ei;
  const int* dstp = ei + nE;

  // workspace layout (128B-aligned offsets)
  char* ws = (char*)d_ws;
  size_t off = 0;
  float* deg  = (float*)(ws + off); off += ((size_t)N * 4 + 127) & ~(size_t)127;
  float* norm = (float*)(ws + off); off += ((size_t)total * 4 + 127) & ~(size_t)127;
  float* bufA = (float*)(ws + off); off += (size_t)N * HID * 4;
  float* bufB = (float*)(ws + off);

  const int NC = N * HID;                   // 12.8M elems
  dim3 blk(256);
  dim3 g_nodes((N + 255) / 256), g_edges((nE + 255) / 256);
  dim3 g_all((total + 255) / 256), g_nc((NC + 255) / 256);
  dim3 g_gemm((N + 63) / 64);
  dim3 g_prop((total + 7) / 8);

  // normalization (once)
  deg_init<<<g_nodes, blk, 0, stream>>>(deg, N);
  deg_acc<<<g_edges, blk, 0, stream>>>(deg, dstp, ew, nE);
  deg_rsqrt<<<g_nodes, blk, 0, stream>>>(deg, N);
  norm_kernel<<<g_all, blk, 0, stream>>>(deg, srcp, dstp, ew, norm, nE, total);

  // layer 1: x(512) -> bufB, aggregate (+b1) into bufA
  gemm_bf16wmma<<<g_gemm, blk, 0, stream>>>(x, W1, bufB, N, K1);
  bias_init<<<g_nc, blk, 0, stream>>>(bufA, b1, NC);
  prop_kernel<<<g_prop, blk, 0, stream>>>(bufB, srcp, dstp, norm, bufA, nE, total);

  // layer 2
  gemm_bf16wmma<<<g_gemm, blk, 0, stream>>>(bufA, W2, bufB, N, HID);
  bias_init<<<g_nc, blk, 0, stream>>>(bufA, b2, NC);
  prop_kernel<<<g_prop, blk, 0, stream>>>(bufB, srcp, dstp, norm, bufA, nE, total);

  // layer 3
  gemm_bf16wmma<<<g_gemm, blk, 0, stream>>>(bufA, W3, bufB, N, HID);
  bias_init<<<g_nc, blk, 0, stream>>>(bufA, b3, NC);
  prop_kernel<<<g_prop, blk, 0, stream>>>(bufB, srcp, dstp, norm, bufA, nE, total);

  // layer 4: aggregate (+b4) directly into d_out, then PReLU in place
  gemm_bf16wmma<<<g_gemm, blk, 0, stream>>>(bufA, W4, bufB, N, HID);
  bias_init<<<g_nc, blk, 0, stream>>>(out, b4, NC);
  prop_kernel<<<g_prop, blk, 0, stream>>>(bufB, srcp, dstp, norm, out, nE, total);
  prelu_kernel<<<g_nc, blk, 0, stream>>>(out, pa, NC);
}
